// LocalAttention_47966194762023
// MI455X (gfx1250) — matrix operand order repaired and checked by reading, not compiled, not run on hardware
//
#include <hip/hip_runtime.h>

#define DIM   256
#define WIN   31
#define BATCH 8
#define SEQ   7936
#define NWIN  (SEQ / WIN)   // 256

typedef __attribute__((ext_vector_type(16))) __bf16 v16bf;
typedef __attribute__((ext_vector_type(8)))  float  v8f;
typedef int v4i_t __attribute__((vector_size(16)));   // matches builtin's b128 pointee

#define SX 264   // row stride (elems) for 32x256 bf16 buffers (x, q, k, att_out)
#define SV 40    // row stride for vT [256][32]
#define SS 36    // row stride for scores (f32) / attn (bf16) [32][32]

// Swizzled-weight geometry: Wqkv -> 48 n-tiles, Wproj -> 16 n-tiles; 8 k-chunks;
// per (nt,kc): 32 lanes x 16 bf16 (exact B-fragment register image).
#define QKV_TRIPLES  (48 * 8 * 32)           // 12288
#define PROJ_TRIPLES (16 * 8 * 32)           // 4096
#define SWZ_BYTES    ((QKV_TRIPLES + PROJ_TRIPLES) * 16 * 2)  // 524288

#if defined(__has_builtin)
#if __has_builtin(__builtin_amdgcn_global_load_async_to_lds_b128)
#define HAVE_ASYNC_LDS 1
#endif
#endif

static __device__ __forceinline__ v8f wmma_bf16(v16bf a, v16bf b, v8f c) {
  return __builtin_amdgcn_wmma_f32_16x16x32_bf16(false, a, false, b, (short)0, c, false, false);
}

static __device__ __forceinline__ void wait_asynccnt0() {
#if defined(__has_builtin) && __has_builtin(__builtin_amdgcn_s_wait_asynccnt)
  __builtin_amdgcn_s_wait_asynccnt(0);
#else
  asm volatile("s_wait_asynccnt 0" ::: "memory");
#endif
}

// A-matrix 16x32 bf16 fragment from LDS (row-major, stride in elems).
static __device__ __forceinline__ v16bf load_a_lds(const __bf16* base, int stride,
                                                   int row0, int kb, int lane) {
  v16bf f;
  const __bf16* rp = base + (row0 + (lane & 15)) * stride + kb;
  const int hi = (lane >> 4) << 4;
#pragma unroll
  for (int p = 0; p < 8; ++p) {
    int ko = hi + (p << 1);
    f[2 * p]     = rp[ko];
    f[2 * p + 1] = rp[ko + 1];
  }
  return f;
}

// B-matrix 32x16 fragment from B^T rows in LDS (lane half selects K 0-15 / 16-31).
static __device__ __forceinline__ v16bf load_b_lds(const __bf16* base, int stride,
                                                   int n0, int kb, int lane) {
  v16bf f;
  const __bf16* rp = base + (n0 + (lane & 15)) * stride + kb + ((lane >> 4) << 4);
#pragma unroll
  for (int j = 0; j < 16; ++j) f[j] = rp[j];
  return f;
}

// B fragment from global pre-swizzled bf16 weights: 32 contiguous bytes per lane.
static __device__ __forceinline__ v16bf load_b_swz(const __bf16* __restrict__ W,
                                                   int nt, int kc, int lane) {
  const __bf16* p = W + ((size_t)((((nt << 3) + kc) << 5) + lane) << 4);
  v16bf f;
#pragma unroll
  for (int j = 0; j < 16; ++j) f[j] = p[j];
  return f;
}

// Fallback: B fragment from global fp32 row-major W[N][K], cvt inline.
static __device__ __forceinline__ v16bf load_b_gf32(const float* __restrict__ W, int ldw,
                                                    int n0, int kb, int lane) {
  v16bf f;
  const float* rp = W + (size_t)(n0 + (lane & 15)) * ldw + kb + ((lane >> 4) << 4);
#pragma unroll
  for (int q = 0; q < 4; ++q) {
    float4 v = *(const float4*)(rp + 4 * q);
    f[4 * q + 0] = (__bf16)v.x;
    f[4 * q + 1] = (__bf16)v.y;
    f[4 * q + 2] = (__bf16)v.z;
    f[4 * q + 3] = (__bf16)v.w;
  }
  return f;
}

// ---- Prepass: fp32 weights -> bf16, pre-swizzled into B-fragment layout ----
__global__ __launch_bounds__(256) void swizzle_weights(
    const float* __restrict__ Wqkv, const float* __restrict__ Wproj,
    __bf16* __restrict__ dq, __bf16* __restrict__ dp) {
  int t = blockIdx.x * 256 + threadIdx.x;   // 0 .. 16383
  const float* W;
  __bf16* D;
  int idx;
  if (t < QKV_TRIPLES) { W = Wqkv;  D = dq; idx = t; }
  else                 { W = Wproj; D = dp; idx = t - QKV_TRIPLES; }
  const int lane = idx & 31, kc = (idx >> 5) & 7, nt = idx >> 8;
  const float* rp = W + (size_t)((nt << 4) + (lane & 15)) * DIM + kc * 32 + ((lane >> 4) << 4);
  __bf16* o = D + ((size_t)idx << 4);
#pragma unroll
  for (int q = 0; q < 4; ++q) {
    float4 v = *(const float4*)(rp + 4 * q);
    o[4 * q + 0] = (__bf16)v.x;
    o[4 * q + 1] = (__bf16)v.y;
    o[4 * q + 2] = (__bf16)v.z;
    o[4 * q + 3] = (__bf16)v.w;
  }
}

template <bool SWZ>
__global__ __launch_bounds__(256) void local_attn_fused(
    const float* __restrict__ x, const float* __restrict__ Wqkv,
    const float* __restrict__ bqkv, const float* __restrict__ Wproj,
    const float* __restrict__ bproj, const __bf16* __restrict__ WqkvS,
    const __bf16* __restrict__ WprojS, float* __restrict__ out) {
  extern __shared__ char smem[];
  __bf16* sXb = (__bf16*)smem;          // [32][SX]  x (bf16), dead after QKV
  __bf16* sQ  = sXb + 32 * SX;          // [32][SX]  q, later reused as att_out
  __bf16* sK  = sQ + 32 * SX;           // [32][SX]  k
  __bf16* sVT = sK + 32 * SX;           // [256][SV] v transposed (channel-major)
  float*  sSc = (float*)smem;                       // [32][SS] f32 scores (overlay sXb)
  __bf16* sAt = (__bf16*)(smem + 32 * SS * 4);      // [32][SS] bf16 attn (overlay sXb)
  __bf16* sAO = sQ;                                 // att_out overlays q

  const int tid  = threadIdx.x;
  const int lane = tid & 31;
  const int w    = tid >> 5;

  const int bidx = blockIdx.x;
  const int win  = bidx & (NWIN - 1);
  const int b    = bidx >> 8;
  const size_t row0g = (size_t)b * SEQ + (size_t)win * WIN;

  // ---- Phase 1: stage x window -> LDS bf16 (pad row 31 zeroed) ----
#if HAVE_ASYNC_LDS
  {
    // Async-copy raw fp32 x (31 rows x 1KB) into the (currently dead) q/k region.
    float* stageF = (float*)(smem + 32 * SX * 2);   // 31744 B < 33792 B available
    for (int c = tid; c < WIN * 16; c += 256) {     // 16B chunks
      int r = c >> 4, c4 = (c & 15) << 2;
      const float* gp = x + (row0g + r) * DIM + c4;
      float* lp = stageF + r * DIM + c4;
      __builtin_amdgcn_global_load_async_to_lds_b128(
          (__attribute__((address_space(1))) v4i_t*)(const void*)gp,
          (__attribute__((address_space(3))) v4i_t*)(void*)lp, 0, 0);
    }
    wait_asynccnt0();
    __syncthreads();
    for (int i = tid; i < 32 * 64; i += 256) {
      int r = i >> 6, c4 = (i & 63) << 2;
      float4 v = make_float4(0.f, 0.f, 0.f, 0.f);
      if (r < WIN) v = *(const float4*)(stageF + r * DIM + c4);
      __bf16* d = sXb + r * SX + c4;
      d[0] = (__bf16)v.x; d[1] = (__bf16)v.y; d[2] = (__bf16)v.z; d[3] = (__bf16)v.w;
    }
  }
#else
  for (int i = tid; i < 32 * 64; i += 256) {
    int r = i >> 6, c4 = (i & 63) << 2;
    float4 v = make_float4(0.f, 0.f, 0.f, 0.f);
    if (r < WIN) v = *(const float4*)(x + (row0g + r) * DIM + c4);
    __bf16* d = sXb + r * SX + c4;
    d[0] = (__bf16)v.x; d[1] = (__bf16)v.y; d[2] = (__bf16)v.z; d[3] = (__bf16)v.w;
  }
#endif
  __syncthreads();

  // ---- Phase 2: qkv = x @ Wqkv^T + bqkv ; wave w owns n-tiles nt = w + 8i ----
  for (int i = 0; i < 6; ++i) {
    const int nt  = w + 8 * i;
    const int gn0 = nt << 4;
    if constexpr (SWZ) {
      if (i < 5)  // prefetch next n-tile's weight block into L2/WGP$
        __builtin_prefetch(WqkvS + ((size_t)((nt + 8) << 3) << 9), 0, 1);
    }
    v8f acc0 = {}, acc1 = {};
#pragma unroll
    for (int kc = 0; kc < 8; ++kc) {
      v16bf bf;
      if constexpr (SWZ) bf = load_b_swz(WqkvS, nt, kc, lane);
      else               bf = load_b_gf32(Wqkv, DIM, gn0, kc * 32, lane);
      v16bf a0 = load_a_lds(sXb, SX, 0, kc * 32, lane);
      acc0 = wmma_bf16(a0, bf, acc0);
      v16bf a1 = load_a_lds(sXb, SX, 16, kc * 32, lane);
      acc1 = wmma_bf16(a1, bf, acc1);
    }
    const int gn    = gn0 + (lane & 15);
    const float bia = bqkv[gn];
    const int rbase = (lane >> 4) << 3;
    if (nt < 16) {
      __bf16* dst = sQ + gn;
#pragma unroll
      for (int g = 0; g < 8; ++g) {
        dst[(rbase + g) * SX]      = (__bf16)(acc0[g] + bia);
        dst[(rbase + g + 16) * SX] = (__bf16)(acc1[g] + bia);
      }
    } else if (nt < 32) {
      __bf16* dst = sK + (gn - 256);
#pragma unroll
      for (int g = 0; g < 8; ++g) {
        dst[(rbase + g) * SX]      = (__bf16)(acc0[g] + bia);
        dst[(rbase + g + 16) * SX] = (__bf16)(acc1[g] + bia);
      }
    } else {
      __bf16* dst = sVT + (size_t)(gn - 512) * SV;   // transposed [chan][token]
#pragma unroll
      for (int g = 0; g < 8; ++g) {
        dst[rbase + g]      = (__bf16)(acc0[g] + bia);
        dst[rbase + g + 16] = (__bf16)(acc1[g] + bia);
      }
    }
  }
  __syncthreads();

  // ---- Phase 3: scores = (q @ k^T) / 16 ; waves 0..3, one 16x16 tile each ----
  if (w < 4) {
    const int mt = w >> 1, nt = w & 1;
    v8f acc = {};
#pragma unroll
    for (int kc = 0; kc < 8; ++kc) {
      v16bf aq = load_a_lds(sQ, SX, mt * 16, kc * 32, lane);
      v16bf bk = load_b_lds(sK, SX, nt * 16, kc * 32, lane);
      acc = wmma_bf16(aq, bk, acc);
    }
    const int col   = nt * 16 + (lane & 15);
    const int rbase = mt * 16 + ((lane >> 4) << 3);
#pragma unroll
    for (int g = 0; g < 8; ++g) sSc[(rbase + g) * SS + col] = acc[g] * 0.0625f;
  }
  __syncthreads();

  // ---- Phase 4: row softmax over 31 valid keys -> bf16 attn ----
  if (tid < 32) {
    float mx = -1e30f;
    for (int j = 0; j < WIN; ++j) mx = fmaxf(mx, sSc[tid * SS + j]);
    float sum = 0.f;
    for (int j = 0; j < WIN; ++j) {
      float e = __expf(sSc[tid * SS + j] - mx);
      sSc[tid * SS + j] = e;
      sum += e;
    }
    float inv = 1.f / sum;
    for (int j = 0; j < WIN; ++j) sAt[tid * SS + j] = (__bf16)(sSc[tid * SS + j] * inv);
    sAt[tid * SS + WIN] = (__bf16)0.f;
  }
  __syncthreads();

  // ---- Phase 5: att_out = attn @ v ; K=32 -> one WMMA per tile ----
  {
#pragma unroll
    for (int t = 0; t < 4; ++t) {
      const int mt = t & 1;
      const int nt = 2 * w + (t >> 1);
      v16bf aa = load_a_lds(sAt, SS, mt * 16, 0, lane);
      v16bf bv = load_b_lds(sVT, SV, nt * 16, 0, lane);
      v8f c = {};
      c = wmma_bf16(aa, bv, c);
      const int col   = nt * 16 + (lane & 15);
      const int rbase = mt * 16 + ((lane >> 4) << 3);
#pragma unroll
      for (int g = 0; g < 8; ++g) sAO[(rbase + g) * SX + col] = (__bf16)c[g];
    }
  }
  __syncthreads();

  // ---- Phase 6: out = att_out @ Wproj^T + bproj -> global (rows 0..30) ----
  for (int t = 0; t < 2; ++t) {
    const int nt = 2 * w + t;
    const int gn = nt * 16 + (lane & 15);
    v8f acc0 = {}, acc1 = {};
#pragma unroll
    for (int kc = 0; kc < 8; ++kc) {
      v16bf bf;
      if constexpr (SWZ) bf = load_b_swz(WprojS, nt, kc, lane);
      else               bf = load_b_gf32(Wproj, DIM, nt * 16, kc * 32, lane);
      v16bf a0 = load_a_lds(sAO, SX, 0, kc * 32, lane);
      acc0 = wmma_bf16(a0, bf, acc0);
      v16bf a1 = load_a_lds(sAO, SX, 16, kc * 32, lane);
      acc1 = wmma_bf16(a1, bf, acc1);
    }
    const float bia = bproj[gn];
    const int rb = (lane >> 4) << 3;
#pragma unroll
    for (int g = 0; g < 8; ++g) {
      int r0 = rb + g;
      if (r0 < WIN) out[(row0g + r0) * DIM + gn] = acc0[g] + bia;
      int r1 = r0 + 16;
      if (r1 < WIN) out[(row0g + r1) * DIM + gn] = acc1[g] + bia;
    }
  }
}

extern "C" void kernel_launch(void* const* d_in, const int* in_sizes, int n_in,
                              void* d_out, int out_size, void* d_ws, size_t ws_size,
                              hipStream_t stream) {
  (void)in_sizes; (void)n_in; (void)out_size;
  const float* x     = (const float*)d_in[0];
  const float* Wqkv  = (const float*)d_in[1];
  const float* bqkv  = (const float*)d_in[2];
  const float* Wproj = (const float*)d_in[3];
  const float* bproj = (const float*)d_in[4];
  float* out = (float*)d_out;

  const size_t smem_bytes = (size_t)(3 * 32 * SX + 256 * SV) * 2;  // 71,168 B

  if (d_ws != nullptr && ws_size >= (size_t)SWZ_BYTES) {
    __bf16* wsQ = (__bf16*)d_ws;
    __bf16* wsP = wsQ + (size_t)QKV_TRIPLES * 16;
    swizzle_weights<<<dim3((QKV_TRIPLES + PROJ_TRIPLES) / 256), dim3(256), 0, stream>>>(
        Wqkv, Wproj, wsQ, wsP);
    local_attn_fused<true><<<dim3(BATCH * NWIN), dim3(256), smem_bytes, stream>>>(
        x, Wqkv, bqkv, Wproj, bproj, wsQ, wsP, out);
  } else {
    local_attn_fused<false><<<dim3(BATCH * NWIN), dim3(256), smem_bytes, stream>>>(
        x, Wqkv, bqkv, Wproj, bproj, nullptr, nullptr, out);
  }
}
